// MultiHeadSelfAttention_81527069213231
// MI455X (gfx1250) — matrix operand order repaired and checked by reading, not compiled, not run on hardware
//
#include <hip/hip_runtime.h>

typedef __attribute__((ext_vector_type(16))) _Float16 v16h;
typedef __attribute__((ext_vector_type(8)))  float    v8f;

#define HID   768
#define NH    12
#define HD    64
#define BB    2
#define TT    2048
#define MTOT  (BB*TT)        // 4096 rows of x
#define NQKV  (3*HID)        // 2304
#define ATTN_SCALE 0.125f    // 1/sqrt(64)

// A-fragment K-offset pattern for 16-bit WMMA A (16x32), matching B:
// lanes 0-15 hold K=0..15 (vgpr v -> 2v,2v+1), lanes 16-31 hold K=16..31
__device__ __forceinline__ int a_koff(int v, int half) {
  return 2 * v + 16 * half;
}
// B-fragment K-offset pattern for 16-bit WMMA B (32x16):
// lanes 0-15 hold K=0..15 (vgpr v -> 2v,2v+1), lanes 16-31 hold K=16..31
__device__ __forceinline__ int b_koff(int v, int half) {
  return 2 * v + 16 * half;
}

// ---------------------------------------------------------------------------
// Kernel 1: QKV projection.  C[M=4096, N=2304] = x[M,768] * w_qkv^T + b_qkv
// One 16x64 output strip per wave (4 WMMAs / k-step, A reused 4x).
// The 64-wide strip maps to exactly one head of Q, K or V.
// Q,K stored f16 [B,H,T,D]; V stored TRANSPOSED f16 [B,H,D,T] so the
// attention kernel's P*V B-fragments are contiguous 128-bit loads.
// ---------------------------------------------------------------------------
__global__ __launch_bounds__(256) void qkv_gemm_wmma(
    const float* __restrict__ x, const float* __restrict__ w,
    const float* __restrict__ bias, _Float16* __restrict__ qb,
    _Float16* __restrict__ kb, _Float16* __restrict__ vb) {
  const int lane = threadIdx.x & 31;
  const int wid  = blockIdx.x * (blockDim.x >> 5) + (threadIdx.x >> 5);
  const int ntiles = NQKV / 64;                 // 36 strips of 64 columns
  const int mt = wid / ntiles, nt = wid % ntiles;
  if (mt >= MTOT / 16) return;
  const int m0 = mt * 16, n0 = nt * 64;
  const int half = lane >> 4, mn = lane & 15;

  const float* arow  = x + (size_t)(m0 + mn) * HID;        // A row m
  const float* brow0 = w + (size_t)(n0 +  0 + mn) * HID;   // B[k][n] = w[n][k]
  const float* brow1 = w + (size_t)(n0 + 16 + mn) * HID;
  const float* brow2 = w + (size_t)(n0 + 32 + mn) * HID;
  const float* brow3 = w + (size_t)(n0 + 48 + mn) * HID;

  v8f acc0 = {}, acc1 = {}, acc2 = {}, acc3 = {};
  for (int k = 0; k < HID; k += 32) {
    v16h a, b0, b1, b2, b3;
#pragma unroll
    for (int v = 0; v < 8; ++v) {
      const int ka = k + a_koff(v, half);
      a[2 * v]     = (_Float16)arow[ka];
      a[2 * v + 1] = (_Float16)arow[ka + 1];
      const int kk = k + b_koff(v, half);
      b0[2 * v] = (_Float16)brow0[kk]; b0[2 * v + 1] = (_Float16)brow0[kk + 1];
      b1[2 * v] = (_Float16)brow1[kk]; b1[2 * v + 1] = (_Float16)brow1[kk + 1];
      b2[2 * v] = (_Float16)brow2[kk]; b2[2 * v + 1] = (_Float16)brow2[kk + 1];
      b3[2 * v] = (_Float16)brow3[kk]; b3[2 * v + 1] = (_Float16)brow3[kk + 1];
    }
    acc0 = __builtin_amdgcn_wmma_f32_16x16x32_f16(false, a, false, b0, (short)0, acc0, false, false);
    acc1 = __builtin_amdgcn_wmma_f32_16x16x32_f16(false, a, false, b1, (short)0, acc1, false, false);
    acc2 = __builtin_amdgcn_wmma_f32_16x16x32_f16(false, a, false, b2, (short)0, acc2, false, false);
    acc3 = __builtin_amdgcn_wmma_f32_16x16x32_f16(false, a, false, b3, (short)0, acc3, false, false);
  }

  // n0 is a multiple of 64 -> whole strip lies in one of Q/K/V and one head.
  const int which = n0 / HID;            // 0=Q 1=K 2=V  (uniform)
  const int h     = (n0 % HID) / HD;     // head          (uniform)
#pragma unroll
  for (int sub = 0; sub < 4; ++sub) {
    const v8f acc = (sub == 0) ? acc0 : ((sub == 1) ? acc1 : ((sub == 2) ? acc2 : acc3));
    const int d  = sub * 16 + mn;        // 0..63 within the head
    const float bn = bias[n0 + sub * 16 + mn];
#pragma unroll
    for (int r = 0; r < 8; ++r) {
      const int mg = m0 + r + 8 * half;  // C-row (ISA C/D layout)
      const int bi = mg >> 11, t = mg & (TT - 1);
      const _Float16 val = (_Float16)(acc[r] + bn);
      if (which == 0)
        qb[(((size_t)bi * NH + h) * TT + t) * HD + d] = val;
      else if (which == 1)
        kb[(((size_t)bi * NH + h) * TT + t) * HD + d] = val;
      else
        vb[(((size_t)bi * NH + h) * HD + d) * TT + t] = val;   // transposed
    }
  }
}

// ---------------------------------------------------------------------------
// Kernel 2: flash attention.  One wave per (b, h, 16-query tile).
// Loop over 32-key tiles: scores via 4 WMMAs (K = head-dim 64), online
// softmax in fp32 registers (shfl_xor within wave32 half-groups), P through
// a wave-private LDS tile (C-layout -> A-fragment), P*V via 4 WMMAs with
// contiguous B-fragments from the transposed V.
// ---------------------------------------------------------------------------
__global__ __launch_bounds__(256) void attn_wmma(
    const _Float16* __restrict__ qb, const _Float16* __restrict__ kb,
    const _Float16* __restrict__ vb, const int* __restrict__ mask,
    _Float16* __restrict__ ctxb) {
  __shared__ _Float16 ptile[8][16 * 32];       // per-wave P tile (1 KB each)
  const int lane  = threadIdx.x & 31;
  const int wslot = threadIdx.x >> 5;
  const int wid   = blockIdx.x * (blockDim.x >> 5) + wslot;
  const int qtiles = TT / 16;                  // 128
  const int bh = wid / qtiles;                 // 0..23
  const int qt = wid % qtiles;
  const int b  = bh / NH;
  const int hh = bh % NH;
  const int q0 = qt * 16;
  const int half = lane >> 4, mn = lane & 15;

  const _Float16* qbase  = qb + (size_t)bh * TT * HD;   // [T][D]
  const _Float16* kbase  = kb + (size_t)bh * TT * HD;   // [T][D]
  const _Float16* vtbase = vb + (size_t)bh * HD * TT;   // [D][T] (transposed)
  const int* mrow = mask + b * TT;

  // Q A-fragments, loaded once: d = 0..31 and 32..63
  v16h qa0, qa1;
  {
    const _Float16* qrow = qbase + (size_t)(q0 + mn) * HD;
#pragma unroll
    for (int v = 0; v < 8; ++v) {
      const int d0 = a_koff(v, half);
      qa0[2 * v]     = qrow[d0];
      qa0[2 * v + 1] = qrow[d0 + 1];
      qa1[2 * v]     = qrow[32 + d0];
      qa1[2 * v + 1] = qrow[32 + d0 + 1];
    }
  }

  float mi[8], li[8];
  v8f ctx0 = {}, ctx1 = {}, ctx2 = {}, ctx3 = {};
#pragma unroll
  for (int r = 0; r < 8; ++r) { mi[r] = -1e30f; li[r] = 0.0f; }

  for (int k0 = 0; k0 < TT; k0 += 32) {
    if (k0 + 32 < TT) {  // global_prefetch_b8 for next key tile
      __builtin_prefetch(kbase + (size_t)(k0 + 32 + mn) * HD, 0, 1);
      __builtin_prefetch(vtbase + (size_t)mn * TT + k0 + 32, 0, 1);
    }
    // ---- scores: two 16x16 key sub-tiles, K-dim = head-dim 64 ----
    v8f sc0 = {}, sc1 = {};
#pragma unroll
    for (int sub = 0; sub < 2; ++sub) {
      const _Float16* krow = kbase + (size_t)(k0 + sub * 16 + mn) * HD;
      v16h klo, khi;                           // B fragments: d 0..31 / 32..63
#pragma unroll
      for (int v = 0; v < 8; ++v) {
        const int d0 = b_koff(v, half);
        klo[2 * v]     = krow[d0];
        klo[2 * v + 1] = krow[d0 + 1];
        khi[2 * v]     = krow[32 + d0];
        khi[2 * v + 1] = krow[32 + d0 + 1];
      }
      if (sub == 0) {
        sc0 = __builtin_amdgcn_wmma_f32_16x16x32_f16(false, qa0, false, klo,
                                                     (short)0, sc0, false, false);
        sc0 = __builtin_amdgcn_wmma_f32_16x16x32_f16(false, qa1, false, khi,
                                                     (short)0, sc0, false, false);
      } else {
        sc1 = __builtin_amdgcn_wmma_f32_16x16x32_f16(false, qa0, false, klo,
                                                     (short)0, sc1, false, false);
        sc1 = __builtin_amdgcn_wmma_f32_16x16x32_f16(false, qa1, false, khi,
                                                     (short)0, sc1, false, false);
      }
    }
    // ---- key mask (lane mn owns key column k0+mn / k0+16+mn) ----
    const bool ok0 = (mrow[k0 + mn] != 0);
    const bool ok1 = (mrow[k0 + 16 + mn] != 0);
    // ---- online softmax per C-row (m = r + 8*half lives in one half-wave) --
#pragma unroll
    for (int r = 0; r < 8; ++r) {
      const float s0 = ok0 ? sc0[r] * ATTN_SCALE : -1e30f;
      const float s1 = ok1 ? sc1[r] * ATTN_SCALE : -1e30f;
      float rmax = fmaxf(s0, s1);
#pragma unroll
      for (int sh = 1; sh < 16; sh <<= 1)
        rmax = fmaxf(rmax, __shfl_xor(rmax, sh, 32));
      const float mnew  = fmaxf(mi[r], rmax);
      const float alpha = __expf(mi[r] - mnew);
      const float p0 = __expf(s0 - mnew);
      const float p1 = __expf(s1 - mnew);
      float rsum = p0 + p1;
#pragma unroll
      for (int sh = 1; sh < 16; sh <<= 1)
        rsum += __shfl_xor(rsum, sh, 32);
      li[r] = li[r] * alpha + rsum;
      mi[r] = mnew;
      ctx0[r] *= alpha; ctx1[r] *= alpha; ctx2[r] *= alpha; ctx3[r] *= alpha;
      const int m = r + 8 * half;
      ptile[wslot][m * 32 + mn]      = (_Float16)p0;
      ptile[wslot][m * 32 + 16 + mn] = (_Float16)p1;
    }
    __syncthreads();
    // ---- re-read P in A-fragment layout (16 queries x 32 keys) ----
    v16h pa;
#pragma unroll
    for (int v = 0; v < 8; ++v) {
      const int kk = a_koff(v, half);
      pa[2 * v]     = ptile[wslot][mn * 32 + kk];
      pa[2 * v + 1] = ptile[wslot][mn * 32 + kk + 1];
    }
    __syncthreads();
    // ---- ctx += P * V : 4 N-tiles over head-dim, K = 32 keys ----
    // V is [D][T]: each lane's 16 halves are contiguous -> b128 loads.
#pragma unroll
    for (int dt = 0; dt < 4; ++dt) {
      const _Float16* vrow = vtbase + (size_t)(dt * 16 + mn) * TT + k0 + 16 * half;
      v16h vf;
#pragma unroll
      for (int v = 0; v < 8; ++v) {
        vf[2 * v]     = vrow[2 * v];
        vf[2 * v + 1] = vrow[2 * v + 1];
      }
      v8f& cacc = (dt == 0) ? ctx0 : ((dt == 1) ? ctx1 : ((dt == 2) ? ctx2 : ctx3));
      cacc = __builtin_amdgcn_wmma_f32_16x16x32_f16(false, pa, false, vf,
                                                    (short)0, cacc, false, false);
    }
  }
  // ---- epilogue: normalize, store ctx as f16 rows [B*T, 768] -------------
#pragma unroll
  for (int r = 0; r < 8; ++r) {
    const float inv = 1.0f / li[r];
    const int m = r + 8 * half;
    const size_t row = ((size_t)b * TT + q0 + m) * HID + hh * HD;
    ctxb[row + 0 * 16 + mn] = (_Float16)(ctx0[r] * inv);
    ctxb[row + 1 * 16 + mn] = (_Float16)(ctx1[r] * inv);
    ctxb[row + 2 * 16 + mn] = (_Float16)(ctx2[r] * inv);
    ctxb[row + 3 * 16 + mn] = (_Float16)(ctx3[r] * inv);
  }
}

// ---------------------------------------------------------------------------
// Kernel 3: output projection.  out[4096,768] = ctx[4096,768]*w_out^T + b_out
// One 16x64 strip per wave (4 WMMAs / k-step, A reused 4x).
// ---------------------------------------------------------------------------
__global__ __launch_bounds__(256) void out_gemm_wmma(
    const _Float16* __restrict__ ctx, const float* __restrict__ w,
    const float* __restrict__ bias, float* __restrict__ out) {
  const int lane = threadIdx.x & 31;
  const int wid  = blockIdx.x * (blockDim.x >> 5) + (threadIdx.x >> 5);
  const int ntiles = HID / 64;                 // 12 strips
  const int mt = wid / ntiles, nt = wid % ntiles;
  if (mt >= MTOT / 16) return;
  const int m0 = mt * 16, n0 = nt * 64;
  const int half = lane >> 4, mn = lane & 15;

  const _Float16* arow = ctx + (size_t)(m0 + mn) * HID;
  const float* brow0 = w + (size_t)(n0 +  0 + mn) * HID;   // B[k][n] = w[n][k]
  const float* brow1 = w + (size_t)(n0 + 16 + mn) * HID;
  const float* brow2 = w + (size_t)(n0 + 32 + mn) * HID;
  const float* brow3 = w + (size_t)(n0 + 48 + mn) * HID;

  v8f acc0 = {}, acc1 = {}, acc2 = {}, acc3 = {};
  for (int k = 0; k < HID; k += 32) {
    v16h a, b0, b1, b2, b3;
#pragma unroll
    for (int v = 0; v < 8; ++v) {
      const int ka = k + a_koff(v, half);
      a[2 * v]     = arow[ka];
      a[2 * v + 1] = arow[ka + 1];
      const int kk = k + b_koff(v, half);
      b0[2 * v] = (_Float16)brow0[kk]; b0[2 * v + 1] = (_Float16)brow0[kk + 1];
      b1[2 * v] = (_Float16)brow1[kk]; b1[2 * v + 1] = (_Float16)brow1[kk + 1];
      b2[2 * v] = (_Float16)brow2[kk]; b2[2 * v + 1] = (_Float16)brow2[kk + 1];
      b3[2 * v] = (_Float16)brow3[kk]; b3[2 * v + 1] = (_Float16)brow3[kk + 1];
    }
    acc0 = __builtin_amdgcn_wmma_f32_16x16x32_f16(false, a, false, b0, (short)0, acc0, false, false);
    acc1 = __builtin_amdgcn_wmma_f32_16x16x32_f16(false, a, false, b1, (short)0, acc1, false, false);
    acc2 = __builtin_amdgcn_wmma_f32_16x16x32_f16(false, a, false, b2, (short)0, acc2, false, false);
    acc3 = __builtin_amdgcn_wmma_f32_16x16x32_f16(false, a, false, b3, (short)0, acc3, false, false);
  }
#pragma unroll
  for (int sub = 0; sub < 4; ++sub) {
    const v8f acc = (sub == 0) ? acc0 : ((sub == 1) ? acc1 : ((sub == 2) ? acc2 : acc3));
    const float bn = bias[n0 + sub * 16 + mn];
#pragma unroll
    for (int r = 0; r < 8; ++r) {
      const int mg = m0 + r + 8 * half;
      out[(size_t)mg * HID + n0 + sub * 16 + mn] = acc[r] + bn;
    }
  }
}

// ---------------------------------------------------------------------------
extern "C" void kernel_launch(void* const* d_in, const int* in_sizes, int n_in,
                              void* d_out, int out_size, void* d_ws, size_t ws_size,
                              hipStream_t stream) {
  const float* x     = (const float*)d_in[0];
  const int*   mask  = (const int*)d_in[1];
  const float* w_qkv = (const float*)d_in[2];
  const float* b_qkv = (const float*)d_in[3];
  const float* w_out = (const float*)d_in[4];
  const float* b_out = (const float*)d_in[5];
  float* out = (float*)d_out;

  const size_t nBHTD = (size_t)BB * NH * TT * HD;  // 3,145,728 elems
  _Float16* qb  = (_Float16*)d_ws;
  _Float16* kb  = qb + nBHTD;
  _Float16* vb  = kb + nBHTD;                      // stored [B,H,D,T]
  _Float16* ctx = vb + nBHTD;                      // total 24 MB of d_ws

  {
    const int tiles  = (MTOT / 16) * (NQKV / 64);  // 9216 waves
    const int blocks = tiles / 8;                  // 8 waves / 256-thread block
    qkv_gemm_wmma<<<blocks, 256, 0, stream>>>(x, w_qkv, b_qkv, qb, kb, vb);
  }
  {
    const int waves  = BB * NH * (TT / 16);        // 3072 waves
    const int blocks = waves / 8;
    attn_wmma<<<blocks, 256, 0, stream>>>(qb, kb, vb, mask, ctx);
  }
  {
    const int tiles  = (MTOT / 16) * (HID / 64);   // 3072 waves
    const int blocks = tiles / 8;
    out_gemm_wmma<<<blocks, 256, 0, stream>>>(ctx, w_out, b_out, out);
  }
}
